// HunyuanMoE_44573170598020
// MI455X (gfx1250) — matrix operand order repaired and checked by reading, not compiled, not run on hardware
//
#include <hip/hip_runtime.h>
#include <hip/hip_bf16.h>

#define T_ 8192
#define H_ 2048
#define F_ 8192
#define E_ 16
#define I_ 1024

typedef __attribute__((ext_vector_type(16))) __bf16   v16bf;
typedef __attribute__((ext_vector_type(8)))  float    v8f;
typedef __attribute__((ext_vector_type(8)))  unsigned int v8u;

union FragB { v8u u; v16bf b; };

__device__ __forceinline__ unsigned short f2bf(float f) {
    union { float f; unsigned int u; } c; c.f = f;
    unsigned int r = c.u + 0x7FFFu + ((c.u >> 16) & 1u);   // round-to-nearest-even
    return (unsigned short)(r >> 16);
}

// pack two fp32 -> one dword of bf16 (lo = even-k element, matching WMMA pairs)
__device__ __forceinline__ unsigned int f2bf_pk(float lo, float hi) {
    union { float f; unsigned int u; } a, b; a.f = lo; b.f = hi;
    unsigned int rl = (a.u + 0x7FFFu + ((a.u >> 16) & 1u)) >> 16;
    unsigned int rh = (b.u + 0x7FFFu + ((b.u >> 16) & 1u)) & 0xFFFF0000u;
    return rl | rh;
}

// LDS byte-offset of a __shared__ pointer (ptrtoint of AS(3) pointer)
__device__ __forceinline__ unsigned lds_off(const void* p) {
    return (unsigned)(unsigned long long)(__attribute__((address_space(3))) const void*)p;
}

// CDNA5 async DMA: copy 32 bytes global -> LDS (2 x B128), ASYNCcnt-tracked
__device__ __forceinline__ void async_copy32(unsigned lds, const void* gptr) {
    asm volatile("global_load_async_to_lds_b128 %0, %1, off\n\t"
                 "global_load_async_to_lds_b128 %0, %1, off offset:16"
                 :: "v"(lds), "v"((unsigned long long)(size_t)gptr)
                 : "memory");
}
__device__ __forceinline__ void wait_async0() {
    asm volatile("s_wait_asynccnt 0x0" ::: "memory");
}

// Workgroup barrier that only drains LDS (DScnt) — does NOT drain global loads,
// so register-staged global prefetches can span a full K-iteration.
__device__ __forceinline__ void wg_barrier_lds() {
    asm volatile("s_wait_dscnt 0x0\n\t"
                 "s_barrier_signal -1\n\t"
                 "s_barrier_wait -1" ::: "memory");
}

// ---------------------------------------------------------------- fp32 -> bf16
__global__ void k_cvt_bf16(const float* __restrict__ in,
                           unsigned short* __restrict__ out, int n4) {
    int i = blockIdx.x * blockDim.x + threadIdx.x;
    int stride = gridDim.x * blockDim.x;
    for (; i < n4; i += stride) {
        float4 f = ((const float4*)in)[i];
        ushort4 o;
        o.x = f2bf(f.x); o.y = f2bf(f.y); o.z = f2bf(f.z); o.w = f2bf(f.w);
        ((ushort4*)out)[i] = o;
    }
}

// ---------------------------------------------------------------- router top-2
__global__ __launch_bounds__(256)
void k_router(const float* __restrict__ x, const float* __restrict__ gw,
              float* __restrict__ dw) {
    const int lane = threadIdx.x & 31;
    const int wave = threadIdx.x >> 5;
    const int t = blockIdx.x * 8 + wave;

    float acc[E_];
#pragma unroll
    for (int e = 0; e < E_; ++e) acc[e] = 0.f;

    const float* xr = x + (size_t)t * H_;
    for (int h = lane; h < H_; h += 32) {
        float xv = xr[h];
        const float* g = gw + (size_t)h * E_;
#pragma unroll
        for (int e = 0; e < E_; ++e) acc[e] += xv * g[e];
    }
#pragma unroll
    for (int off = 16; off; off >>= 1) {
#pragma unroll
        for (int e = 0; e < E_; ++e) acc[e] += __shfl_xor(acc[e], off, 32);
    }
    if (lane == 0) {
        int i1 = 0; float v1 = acc[0];
#pragma unroll
        for (int e = 1; e < E_; ++e) if (acc[e] > v1) { v1 = acc[e]; i1 = e; }
        int i2 = -1; float v2 = -3.4e38f;
#pragma unroll
        for (int e = 0; e < E_; ++e) if (e != i1 && acc[e] > v2) { v2 = acc[e]; i2 = e; }
        float e2 = __expf(v2 - v1);
        float inv = 1.f / (1.f + e2);
        float* row = dw + (size_t)t * E_;
#pragma unroll
        for (int e = 0; e < E_; ++e) row[e] = 0.f;
        row[i1] = inv;
        row[i2] = e2 * inv;
    }
}

// ------------------------------------------------- fused gate/up WMMA GEMM
// A: bf16 [M,K] ld=K ; Wg,Wu: fp32 [K,N] ld=N ; Hout: bf16 [M,N] = silu(A@Wg)*(A@Wu)
__global__ __launch_bounds__(256)
void k_gateup(const unsigned short* __restrict__ A,
              const float* __restrict__ Wg, const float* __restrict__ Wu,
              unsigned short* __restrict__ Hout, int K, int N) {
    __shared__ unsigned short sA[2][128 * 40];    // ping-pong, filled by async DMA
    __shared__ unsigned short sBg[64 * 36];       // [col][k] pad 36 (transposed)
    __shared__ unsigned short sBu[64 * 36];

    const int tid  = threadIdx.x;
    const int lane = tid & 31;
    const int wave = tid >> 5;
    const int waveM = wave >> 1, waveN = wave & 1;
    const int rowBase = blockIdx.x * 128;
    const int colBase = blockIdx.y * 64;
    const int half = lane >> 4, r15 = lane & 15;

    v8f accG[2][2], accU[2][2];
#pragma unroll
    for (int i = 0; i < 2; ++i)
#pragma unroll
        for (int j = 0; j < 2; ++j) { accG[i][j] = (v8f){}; accU[i][j] = (v8f){}; }

    // A loader: 2 threads/row, 32B (16 bf16) each -> async DMA
    const int arow = tid >> 1, ahalf = tid & 1;
    const unsigned short* Ap = A + (size_t)(rowBase + arow) * K + ahalf * 16;
    const unsigned ldsA0 = lds_off(&sA[0][arow * 40 + ahalf * 16]);
    const unsigned ldsA1 = lds_off(&sA[1][arow * 40 + ahalf * 16]);

    // B loader: each thread owns a k-pair x 4 columns (packed-dword stores)
    const int bkp = (tid & 15) * 2;               // k, k+1
    const int bng = (tid >> 4) * 4;               // 4 consecutive n
    const float* Gp = Wg + (size_t)bkp * N + colBase + bng;
    const float* Up = Wu + (size_t)bkp * N + colBase + bng;

    // prologue: k-tile 0
    async_copy32(ldsA0, Ap);
    float4 rg0 = *(const float4*)Gp, rg1 = *(const float4*)(Gp + N);
    float4 ru0 = *(const float4*)Up, ru1 = *(const float4*)(Up + N);

    int p = 0;
    for (int k0 = 0; k0 < K; k0 += 32, p ^= 1) {
        { // B registers -> LDS (fp32 -> packed bf16 dwords, transposed)
            float g0[4] = {rg0.x, rg0.y, rg0.z, rg0.w};
            float g1[4] = {rg1.x, rg1.y, rg1.z, rg1.w};
            float u0[4] = {ru0.x, ru0.y, ru0.z, ru0.w};
            float u1[4] = {ru1.x, ru1.y, ru1.z, ru1.w};
#pragma unroll
            for (int i = 0; i < 4; ++i) {
                *(unsigned*)&sBg[(bng + i) * 36 + bkp] = f2bf_pk(g0[i], g1[i]);
                *(unsigned*)&sBu[(bng + i) * 36 + bkp] = f2bf_pk(u0[i], u1[i]);
            }
        }
        wait_async0();            // this wave's A slice for buffer p landed
        wg_barrier_lds();         // all waves' A slices + B stores visible

        // pipeline next k-tile: async A -> other buffer, B -> registers
        const int kn = (k0 + 32 < K) ? (k0 + 32) : k0;
        async_copy32(p ? ldsA0 : ldsA1, Ap + kn);
        {
            rg0 = *(const float4*)(Gp + (size_t)kn * N);
            rg1 = *(const float4*)(Gp + (size_t)kn * N + N);
            ru0 = *(const float4*)(Up + (size_t)kn * N);
            ru1 = *(const float4*)(Up + (size_t)kn * N + N);
        }
        if (k0 + 128 < K) {       // stream weights into L2 (global_prefetch_b8)
            __builtin_prefetch(Gp + (size_t)(k0 + 128) * N, 0, 1);
            __builtin_prefetch(Up + (size_t)(k0 + 128) * N, 0, 1);
        }

        FragB aF[2], gF[2], uF[2];
#pragma unroll
        for (int mt = 0; mt < 2; ++mt) {
            const unsigned short* base = &sA[p][(waveM * 32 + mt * 16 + r15) * 40];
#pragma unroll
            for (int v = 0; v < 8; ++v) {
                int kp = (v < 4 ? 2 * v : 8 + 2 * v) + half * 8;   // ISA A-layout
                aF[mt].u[v] = *(const unsigned int*)(base + kp);
            }
        }
#pragma unroll
        for (int nt = 0; nt < 2; ++nt) {
            const unsigned short* bg = sBg + (waveN * 32 + nt * 16 + r15) * 36;
            const unsigned short* bu = sBu + (waveN * 32 + nt * 16 + r15) * 36;
#pragma unroll
            for (int v = 0; v < 8; ++v) {
                int kk = (v < 4 ? 2 * v : 8 + 2 * v) + half * 8;   // ISA B-layout
                gF[nt].u[v] = *(const unsigned int*)(bg + kk);
                uF[nt].u[v] = *(const unsigned int*)(bu + kk);
            }
        }
#pragma unroll
        for (int mt = 0; mt < 2; ++mt)
#pragma unroll
            for (int nt = 0; nt < 2; ++nt) {
                accG[mt][nt] = __builtin_amdgcn_wmma_f32_16x16x32_bf16(
                    false, aF[mt].b, false, gF[nt].b, (short)0, accG[mt][nt], false, false);
                accU[mt][nt] = __builtin_amdgcn_wmma_f32_16x16x32_bf16(
                    false, aF[mt].b, false, uF[nt].b, (short)0, accU[mt][nt], false, false);
            }
        wg_barrier_lds();         // retire sB/sA[p] reads before next overwrite
    }

    // epilogue: h = silu(g) * u = g * sigmoid(g) * u, stored bf16
#pragma unroll
    for (int mt = 0; mt < 2; ++mt)
#pragma unroll
        for (int nt = 0; nt < 2; ++nt) {
#pragma unroll
            for (int r = 0; r < 8; ++r) {
                float g = accG[mt][nt][r];
                float u = accU[mt][nt][r];
                float sig = __builtin_amdgcn_rcpf(1.f + __expf(-g));
                float h = g * sig * u;
                int row = rowBase + waveM * 32 + mt * 16 + half * 8 + r;
                int col = colBase + waveN * 32 + nt * 16 + r15;
                Hout[(size_t)row * N + col] = f2bf(h);
            }
        }
}

// ------------------------------------------------- down-proj WMMA GEMM
// scale==null : Out = A@Wd          (initializes output; shared-expert pass)
// scale!=null : Out += scale[row*E_] * (A@Wd)   (per-expert accumulate)
__global__ __launch_bounds__(256)
void k_down(const unsigned short* __restrict__ A,
            const float* __restrict__ Wd,
            float* __restrict__ Out,
            const float* __restrict__ scale, int K, int N) {
    __shared__ unsigned short sA[2][128 * 40];
    __shared__ unsigned short sB[64 * 36];

    const int tid  = threadIdx.x;
    const int lane = tid & 31;
    const int wave = tid >> 5;
    const int waveM = wave >> 1, waveN = wave & 1;
    const int rowBase = blockIdx.x * 128;
    const int colBase = blockIdx.y * 64;
    const int half = lane >> 4, r15 = lane & 15;

    v8f acc[2][2];
#pragma unroll
    for (int i = 0; i < 2; ++i)
#pragma unroll
        for (int j = 0; j < 2; ++j) acc[i][j] = (v8f){};

    const int arow = tid >> 1, ahalf = tid & 1;
    const unsigned short* Ap = A + (size_t)(rowBase + arow) * K + ahalf * 16;
    const unsigned ldsA0 = lds_off(&sA[0][arow * 40 + ahalf * 16]);
    const unsigned ldsA1 = lds_off(&sA[1][arow * 40 + ahalf * 16]);

    const int bkp = (tid & 15) * 2;
    const int bng = (tid >> 4) * 4;
    const float* Wp = Wd + (size_t)bkp * N + colBase + bng;

    async_copy32(ldsA0, Ap);
    float4 rw0 = *(const float4*)Wp, rw1 = *(const float4*)(Wp + N);

    int p = 0;
    for (int k0 = 0; k0 < K; k0 += 32, p ^= 1) {
        {
            float w0[4] = {rw0.x, rw0.y, rw0.z, rw0.w};
            float w1[4] = {rw1.x, rw1.y, rw1.z, rw1.w};
#pragma unroll
            for (int i = 0; i < 4; ++i)
                *(unsigned*)&sB[(bng + i) * 36 + bkp] = f2bf_pk(w0[i], w1[i]);
        }
        wait_async0();
        wg_barrier_lds();

        const int kn = (k0 + 32 < K) ? (k0 + 32) : k0;
        async_copy32(p ? ldsA0 : ldsA1, Ap + kn);
        rw0 = *(const float4*)(Wp + (size_t)kn * N);
        rw1 = *(const float4*)(Wp + (size_t)kn * N + N);
        if (k0 + 128 < K)
            __builtin_prefetch(Wp + (size_t)(k0 + 128) * N, 0, 1);

        FragB aF[2], bF[2];
#pragma unroll
        for (int mt = 0; mt < 2; ++mt) {
            const unsigned short* base = &sA[p][(waveM * 32 + mt * 16 + r15) * 40];
#pragma unroll
            for (int v = 0; v < 8; ++v) {
                int kp = (v < 4 ? 2 * v : 8 + 2 * v) + half * 8;
                aF[mt].u[v] = *(const unsigned int*)(base + kp);
            }
        }
#pragma unroll
        for (int nt = 0; nt < 2; ++nt) {
            const unsigned short* bb = sB + (waveN * 32 + nt * 16 + r15) * 36;
#pragma unroll
            for (int v = 0; v < 8; ++v) {
                int kk = (v < 4 ? 2 * v : 8 + 2 * v) + half * 8;
                bF[nt].u[v] = *(const unsigned int*)(bb + kk);
            }
        }
#pragma unroll
        for (int mt = 0; mt < 2; ++mt)
#pragma unroll
            for (int nt = 0; nt < 2; ++nt)
                acc[mt][nt] = __builtin_amdgcn_wmma_f32_16x16x32_bf16(
                    false, aF[mt].b, false, bF[nt].b, (short)0, acc[mt][nt], false, false);
        wg_barrier_lds();
    }

    if (scale) {
#pragma unroll
        for (int mt = 0; mt < 2; ++mt)
#pragma unroll
            for (int nt = 0; nt < 2; ++nt)
#pragma unroll
                for (int r = 0; r < 8; ++r) {
                    int row = rowBase + waveM * 32 + mt * 16 + half * 8 + r;
                    int col = colBase + waveN * 32 + nt * 16 + r15;
                    size_t idx = (size_t)row * N + col;
                    Out[idx] += scale[(size_t)row * E_] * acc[mt][nt][r];
                }
    } else {
#pragma unroll
        for (int mt = 0; mt < 2; ++mt)
#pragma unroll
            for (int nt = 0; nt < 2; ++nt)
#pragma unroll
                for (int r = 0; r < 8; ++r) {
                    int row = rowBase + waveM * 32 + mt * 16 + half * 8 + r;
                    int col = colBase + waveN * 32 + nt * 16 + r15;
                    Out[(size_t)row * N + col] = acc[mt][nt][r];
                }
    }
}

// ----------------------------------------------------------------- launcher
extern "C" void kernel_launch(void* const* d_in, const int* in_sizes, int n_in,
                              void* d_out, int out_size, void* d_ws, size_t ws_size,
                              hipStream_t stream) {
    const float* x       = (const float*)d_in[0];
    const float* gate_w  = (const float*)d_in[1];
    const float* w_gate  = (const float*)d_in[2];
    const float* w_up    = (const float*)d_in[3];
    const float* w_down  = (const float*)d_in[4];
    const float* sw_gate = (const float*)d_in[5];
    const float* sw_up   = (const float*)d_in[6];
    const float* sw_down = (const float*)d_in[7];
    float* out = (float*)d_out;

    char* ws = (char*)d_ws;
    unsigned short* xb  = (unsigned short*)ws; ws += (size_t)T_ * H_ * 2;  // x bf16
    unsigned short* hsh = (unsigned short*)ws; ws += (size_t)T_ * F_ * 2;  // shared hidden
    unsigned short* hex = (unsigned short*)ws; ws += (size_t)T_ * I_ * 2;  // expert hidden
    float* densew       = (float*)ws;          ws += (size_t)T_ * E_ * 4;  // routing [T,E]

    k_cvt_bf16<<<4096, 256, 0, stream>>>(x, xb, (T_ * H_) / 4);
    k_router<<<T_ / 8, 256, 0, stream>>>(x, gate_w, densew);

    // shared expert: init d_out
    k_gateup<<<dim3(T_ / 128, F_ / 64), 256, 0, stream>>>(xb, sw_gate, sw_up, hsh, H_, F_);
    k_down<<<dim3(T_ / 128, H_ / 64), 256, 0, stream>>>(hsh, sw_down, out, nullptr, F_, H_);

    // routed experts: accumulate with per-token routing weights
    for (int e = 0; e < E_; ++e) {
        k_gateup<<<dim3(T_ / 128, I_ / 64), 256, 0, stream>>>(
            xb, w_gate + (size_t)e * H_ * I_, w_up + (size_t)e * H_ * I_, hex, H_, I_);
        k_down<<<dim3(T_ / 128, H_ / 64), 256, 0, stream>>>(
            hex, w_down + (size_t)e * I_ * H_, out, densew + e, I_, H_);
    }
}
